// SpikingMLP_55697135895208
// MI455X (gfx1250) — matrix operand order repaired and checked by reading, not compiled, not run on hardware
//
#include <hip/hip_runtime.h>
#include <cstdint>
#include <cstddef>

typedef _Float16 v16h __attribute__((ext_vector_type(16)));
typedef _Float16 v8h  __attribute__((ext_vector_type(8)));
typedef _Float16 v4h  __attribute__((ext_vector_type(4)));
typedef float    v8f  __attribute__((ext_vector_type(8)));
typedef float    v4f  __attribute__((ext_vector_type(4)));

static constexpr int T_STEPS = 8;
static constexpr int BATCH   = 256;
static constexpr int D_IN    = 2048;
static constexpr int D_H     = 4096;
static constexpr int D_OUT   = 1024;

// ---------------------------------------------------------------- utilities
// 4-wide f32 -> f16 conversion (all our sizes are multiples of 4)
__global__ void f32_to_f16_kernel(const float* __restrict__ in,
                                  _Float16* __restrict__ out, int n4) {
  int i = blockIdx.x * blockDim.x + threadIdx.x;
  if (i < n4) {
    v4f x = *reinterpret_cast<const v4f*>(in + (size_t)i * 4);
    v4h y;
    y.x = (_Float16)x.x; y.y = (_Float16)x.y;
    y.z = (_Float16)x.z; y.w = (_Float16)x.w;
    *reinterpret_cast<v4h*>(out + (size_t)i * 4) = y;
  }
}

__global__ void zero_f32_kernel(float* __restrict__ p, int n) {
  int i = blockIdx.x * blockDim.x + threadIdx.x;
  if (i < n) p[i] = 0.0f;
}

// LIF for layer 1: v = (v + h)/2 ; s = (v >= 1) ; v = s ? 0 : v
__global__ void lif_kernel(const float* __restrict__ h,
                           float* __restrict__ v,
                           _Float16* __restrict__ s, int n) {
  int i = blockIdx.x * blockDim.x + threadIdx.x;
  if (i < n) {
    float vv = 0.5f * (v[i] + h[i]);
    bool sp = vv >= 1.0f;
    v[i] = sp ? 0.0f : vv;
    s[i] = (_Float16)(sp ? 1.0f : 0.0f);
  }
}

// ---------------------------------------------------------------- WMMA GEMM
// C[M,N] = A[M,K] * W[N,K]^T + bias, row-major, K contiguous for both.
// Block: 128 threads = 4 waves (2x2); block tile 128x128; wave tile 64x64
// (4 M-frags x 4 N-frags of 16x16 -> 16 WMMA per k-tile vs 16 b128 loads,
//  every fragment reused 4x: ~32 FLOP per cache byte).
//
// MODE 0: out[row*N+col] = h                                  (fc1 -> h1)
// MODE 1: LIF fused, write f16 spikes + update vstate         (fc2 -> s2)
// MODE 2: LIF fused, write f32 spikes to out + update vstate  (fc3 -> d_out)
template <int MODE>
__global__ __launch_bounds__(128)
void gemm_wmma_kernel(const _Float16* __restrict__ A,
                      const _Float16* __restrict__ W,
                      const float* __restrict__ bias,
                      int K, int N,
                      float* __restrict__ outF,
                      float* __restrict__ vstate,
                      _Float16* __restrict__ spikeOut) {
  const int tid  = threadIdx.x;
  const int lane = tid & 31;
  const int wave = tid >> 5;       // 0..3
  const int r    = lane & 15;      // row/col within fragment
  const int g    = lane >> 4;      // half-wave group

  const int m0 = blockIdx.y * 128 + (wave >> 1) * 64;   // 2 wave-rows
  const int n0 = blockIdx.x * 128 + (wave & 1) * 64;    // 2 wave-cols

  const _Float16* aRow[4];
#pragma unroll
  for (int i = 0; i < 4; ++i)
    aRow[i] = A + (size_t)(m0 + i * 16 + r) * (size_t)K;
  const _Float16* bRow[4];
#pragma unroll
  for (int j = 0; j < 4; ++j)
    bRow[j] = W + (size_t)(n0 + j * 16 + r) * (size_t)K;

  v8f acc[4][4] = {};

  for (int k0 = 0; k0 < K; k0 += 32) {
    // Prefetch ahead (speculative; dropped if OOB). Locality 3 -> near/HT.
#pragma unroll
    for (int j = 0; j < 4; ++j)
      __builtin_prefetch(bRow[j] + k0 + 64 + g * 16, 0, 3);
#pragma unroll
    for (int i = 0; i < 4; ++i)
      __builtin_prefetch(aRow[i] + k0 + 64 + g * 8, 0, 3);

    // A fragment: lanes 0-15 hold K {0..15}, lanes 16-31 {16..31}
    union { v16h v; v8h h[2]; } a[4];
#pragma unroll
    for (int i = 0; i < 4; ++i) {
      a[i].h[0] = *reinterpret_cast<const v8h*>(aRow[i] + k0 + g * 16);
      a[i].h[1] = *reinterpret_cast<const v8h*>(aRow[i] + k0 + g * 16 + 8);
    }
    // B fragment: lane = column, 16 contiguous K halves (g selects K half)
    v16h b[4];
#pragma unroll
    for (int j = 0; j < 4; ++j)
      b[j] = *reinterpret_cast<const v16h*>(bRow[j] + k0 + g * 16);

#pragma unroll
    for (int i = 0; i < 4; ++i)
#pragma unroll
      for (int j = 0; j < 4; ++j)
        acc[i][j] = __builtin_amdgcn_wmma_f32_16x16x32_f16(
            false, a[i].v, false, b[j], (short)0, acc[i][j], false, false);
  }

  // Epilogue: C frag element p -> (row = m0+i*16+g*8+p, col = n0+j*16+r)
#pragma unroll
  for (int i = 0; i < 4; ++i) {
#pragma unroll
    for (int j = 0; j < 4; ++j) {
      const int col = n0 + j * 16 + r;
      const float bia = bias[col];
#pragma unroll
      for (int p = 0; p < 8; ++p) {
        const int row = m0 + i * 16 + g * 8 + p;
        const size_t idx = (size_t)row * (size_t)N + (size_t)col;
        float h = acc[i][j][p] + bia;
        if (MODE == 0) {
          outF[idx] = h;
        } else {
          float vv = 0.5f * (vstate[idx] + h);
          bool sp = vv >= 1.0f;
          vstate[idx] = sp ? 0.0f : vv;
          if (MODE == 1)
            spikeOut[idx] = (_Float16)(sp ? 1.0f : 0.0f);
          else
            outF[idx] = sp ? 1.0f : 0.0f;
        }
      }
    }
  }
}

// ---------------------------------------------------------------- launcher
extern "C" void kernel_launch(void* const* d_in, const int* in_sizes, int n_in,
                              void* d_out, int out_size, void* d_ws, size_t ws_size,
                              hipStream_t stream) {
  const float* x  = (const float*)d_in[0];  // [T,B,Din]
  const float* W1 = (const float*)d_in[1];  // [Dh,Din]
  const float* b1 = (const float*)d_in[2];  // [Dh]
  const float* W2 = (const float*)d_in[3];  // [Dh,Dh]
  const float* b2 = (const float*)d_in[4];  // [Dh]
  const float* W3 = (const float*)d_in[5];  // [Dout,Dh]
  const float* b3 = (const float*)d_in[6];  // [Dout]
  float* out = (float*)d_out;               // [T,B,Dout]

  uint8_t* ws = (uint8_t*)d_ws;
  size_t off = 0;
  auto carve = [&](size_t bytes) {
    void* p = ws + off;
    off += (bytes + 255) & ~(size_t)255;
    return p;
  };
  const size_t nX  = (size_t)T_STEPS * BATCH * D_IN;
  const size_t nW1 = (size_t)D_H * D_IN;
  const size_t nW2 = (size_t)D_H * D_H;
  const size_t nW3 = (size_t)D_OUT * D_H;
  const size_t nH1 = (size_t)T_STEPS * BATCH * D_H;

  _Float16* x16 = (_Float16*)carve(nX * 2);
  _Float16* w1h = (_Float16*)carve(nW1 * 2);
  _Float16* w2h = (_Float16*)carve(nW2 * 2);
  _Float16* w3h = (_Float16*)carve(nW3 * 2);
  float*    h1  = (float*)carve(nH1 * 4);
  // v1, v2, v3 carved contiguously so a single zero pass covers them
  float*    v1  = (float*)carve((size_t)BATCH * D_H * 4);
  float*    v2  = (float*)carve((size_t)BATCH * D_H * 4);
  float*    v3  = (float*)carve((size_t)BATCH * D_OUT * 4);
  _Float16* s1  = (_Float16*)carve((size_t)BATCH * D_H * 2);
  _Float16* s2  = (_Float16*)carve((size_t)BATCH * D_H * 2);
  (void)ws_size; (void)in_sizes; (void)n_in; (void)out_size;

  const int TPB = 256;
  auto blocks = [&](size_t n) { return (unsigned)((n + TPB - 1) / TPB); };

  // Zero membrane potentials (v1,v2,v3 are contiguous 256-aligned carves).
  const int nZero = BATCH * D_H + BATCH * D_H + BATCH * D_OUT;
  zero_f32_kernel<<<blocks(nZero), TPB, 0, stream>>>(v1, nZero);

  // f32 -> f16 conversions (4 elements per thread)
  f32_to_f16_kernel<<<blocks(nX / 4),  TPB, 0, stream>>>(x,  x16, (int)(nX / 4));
  f32_to_f16_kernel<<<blocks(nW1 / 4), TPB, 0, stream>>>(W1, w1h, (int)(nW1 / 4));
  f32_to_f16_kernel<<<blocks(nW2 / 4), TPB, 0, stream>>>(W2, w2h, (int)(nW2 / 4));
  f32_to_f16_kernel<<<blocks(nW3 / 4), TPB, 0, stream>>>(W3, w3h, (int)(nW3 / 4));

  const int GEMM_TPB = 128;  // 4 waves, 2x2 of 64x64 wave tiles

  // fc1 batched over all timesteps: [2048 x 2048] @ [2048 x 4096]^T
  {
    dim3 grid(D_H / 128, (T_STEPS * BATCH) / 128);
    gemm_wmma_kernel<0><<<grid, GEMM_TPB, 0, stream>>>(
        x16, w1h, b1, D_IN, D_H, h1, nullptr, nullptr);
  }

  // Sequential time scan
  for (int t = 0; t < T_STEPS; ++t) {
    const int nL1 = BATCH * D_H;
    lif_kernel<<<blocks(nL1), TPB, 0, stream>>>(
        h1 + (size_t)t * BATCH * D_H, v1, s1, nL1);

    // fc2 + LIF2 fused: s2 = LIF(s1 @ W2^T + b2)
    {
      dim3 grid(D_H / 128, BATCH / 128);
      gemm_wmma_kernel<1><<<grid, GEMM_TPB, 0, stream>>>(
          s1, w2h, b2, D_H, D_H, nullptr, v2, s2);
    }
    // fc3 + LIF3 fused: out[t] = LIF(s2 @ W3^T + b3)
    {
      dim3 grid(D_OUT / 128, BATCH / 128);
      gemm_wmma_kernel<2><<<grid, GEMM_TPB, 0, stream>>>(
          s2, w3h, b3, D_H, D_OUT, out + (size_t)t * BATCH * D_OUT, v3, nullptr);
    }
  }
}
